// CausalSelfAttention_90202903150923
// MI455X (gfx1250) — matrix operand order repaired and checked by reading, not compiled, not run on hardware
//
#include <hip/hip_runtime.h>

#define L2E 1.44269504088896340736f

typedef __attribute__((ext_vector_type(16))) __bf16 v16bf;
typedef __attribute__((ext_vector_type(8)))  __bf16 v8bf;
typedef __attribute__((ext_vector_type(8)))  float  v8f;

static __device__ inline v16bf make16(v8bf lo, v8bf hi) {
  v16bf r;
#pragma unroll
  for (int i = 0; i < 8; i++) { r[i] = lo[i]; r[i + 8] = hi[i]; }
  return r;
}

// A fragment (16x32 bf16, MxK) from row-major src [rows, ld]; rows row0..+15, k k0..+31.
// lane: row = row0 + (lane&15); contiguous k = k0 + (lane>=16?16:0) + 0..15
static __device__ inline v16bf load_a(const __bf16* __restrict__ base, int ld,
                                      int row0, int k0, int lane) {
  int row = row0 + (lane & 15);
  int h8  = (lane >> 4) << 4;
  const __bf16* p = base + (size_t)row * ld + k0 + h8;
  return make16(*(const v8bf*)p, *(const v8bf*)(p + 8));
}

// B fragment (32x16 bf16, KxN) where B[k][n] = src[n0+n][k0+k], src row-major [., ld].
// lane: col n = n0 + (lane&15); contiguous k = k0 + (lane>=16?16:0) + 0..15
static __device__ inline v16bf load_bt(const __bf16* __restrict__ src, int ld,
                                       int n0, int k0, int lane) {
  int n  = n0 + (lane & 15);
  int kb = (lane >> 4) << 4;
  const __bf16* p = src + (size_t)n * ld + k0 + kb;
  return make16(*(const v8bf*)p, *(const v8bf*)(p + 8));
}

static __device__ inline v8f wmma_bf16(v16bf a, v16bf b, v8f c) {
  return __builtin_amdgcn_wmma_f32_16x16x32_bf16(false, a, false, b, (short)0, c,
                                                 false, false);
}

__global__ __launch_bounds__(256) void cvt_f32_bf16_v8(const float* __restrict__ src,
                                                       __bf16* __restrict__ dst, int n8) {
  int i = blockIdx.x * blockDim.x + threadIdx.x;
  int stride = gridDim.x * blockDim.x;
  for (; i < n8; i += stride) {
    const float4* s = (const float4*)(src + (size_t)i * 8);
    float4 x0 = s[0], x1 = s[1];
    v8bf r;
    r[0] = (__bf16)x0.x; r[1] = (__bf16)x0.y; r[2] = (__bf16)x0.z; r[3] = (__bf16)x0.w;
    r[4] = (__bf16)x1.x; r[5] = (__bf16)x1.y; r[6] = (__bf16)x1.z; r[7] = (__bf16)x1.w;
    *(v8bf*)(dst + (size_t)i * 8) = r;
  }
}

// Software-pipelined 32x64 tile GEMM core: acc[mi*4+t] for rows m0+mi*16, cols n0+t*16.
// K depth fixed at 1024. Loads for step k are issued before WMMAs of step k-32.
static __device__ inline void gemm_core(const __bf16* __restrict__ A,
                                        const __bf16* __restrict__ W,
                                        int m0, int n0, int lane, v8f acc[8]) {
  v16bf a0 = load_a(A, 1024, m0, 0, lane);
  v16bf a1 = load_a(A, 1024, m0 + 16, 0, lane);
  v16bf b0 = load_bt(W, 1024, n0, 0, lane);
  v16bf b1 = load_bt(W, 1024, n0 + 16, 0, lane);
  v16bf b2 = load_bt(W, 1024, n0 + 32, 0, lane);
  v16bf b3 = load_bt(W, 1024, n0 + 48, 0, lane);
  for (int k0 = 32; k0 < 1024; k0 += 32) {
    // prefetch next k-step while current WMMAs execute
    v16bf na0 = load_a(A, 1024, m0, k0, lane);
    v16bf na1 = load_a(A, 1024, m0 + 16, k0, lane);
    v16bf nb0 = load_bt(W, 1024, n0, k0, lane);
    v16bf nb1 = load_bt(W, 1024, n0 + 16, k0, lane);
    v16bf nb2 = load_bt(W, 1024, n0 + 32, k0, lane);
    v16bf nb3 = load_bt(W, 1024, n0 + 48, k0, lane);
    acc[0] = wmma_bf16(a0, b0, acc[0]);
    acc[1] = wmma_bf16(a0, b1, acc[1]);
    acc[2] = wmma_bf16(a0, b2, acc[2]);
    acc[3] = wmma_bf16(a0, b3, acc[3]);
    acc[4] = wmma_bf16(a1, b0, acc[4]);
    acc[5] = wmma_bf16(a1, b1, acc[5]);
    acc[6] = wmma_bf16(a1, b2, acc[6]);
    acc[7] = wmma_bf16(a1, b3, acc[7]);
    a0 = na0; a1 = na1; b0 = nb0; b1 = nb1; b2 = nb2; b3 = nb3;
  }
  acc[0] = wmma_bf16(a0, b0, acc[0]);
  acc[1] = wmma_bf16(a0, b1, acc[1]);
  acc[2] = wmma_bf16(a0, b2, acc[2]);
  acc[3] = wmma_bf16(a0, b3, acc[3]);
  acc[4] = wmma_bf16(a1, b0, acc[4]);
  acc[5] = wmma_bf16(a1, b1, acc[5]);
  acc[6] = wmma_bf16(a1, b2, acc[6]);
  acc[7] = wmma_bf16(a1, b3, acc[7]);
}

// qkv = xb @ wb^T + bias ; scatter into Q [B,H,L,64], K [B,H,L,64], Vt [B,H,64,L] (bf16)
__global__ __launch_bounds__(256) void qkv_gemm(const __bf16* __restrict__ xb,
                                                const __bf16* __restrict__ wb,
                                                const float* __restrict__ bias,
                                                __bf16* __restrict__ Q,
                                                __bf16* __restrict__ K,
                                                __bf16* __restrict__ Vt) {
  const int lane = threadIdx.x & 31;
  const int wid  = (blockIdx.x * blockDim.x + threadIdx.x) >> 5;
  const int NG = 3072 / 64;
  const int m0 = (wid / NG) * 32;
  const int n0 = (wid % NG) * 64;
  v8f acc[8] = {};
  gemm_core(xb, wb, m0, n0, lane, acc);

  const int h8 = (lane >> 4) << 3;
#pragma unroll
  for (int mi = 0; mi < 2; mi++) {
#pragma unroll
    for (int t = 0; t < 4; t++) {
      int n = n0 + t * 16 + (lane & 15);
      float bv = bias[n];
      if (n0 + t * 16 >= 2048) {
        // V block: rows l are consecutive per lane -> pack one b128 store into Vt
        int c = n - 2048, h = c >> 6, d = c & 63;
        int r0 = m0 + mi * 16 + h8;
        int b = r0 >> 11, l0 = r0 & 2047;
        v8bf r;
#pragma unroll
        for (int v = 0; v < 8; v++) r[v] = (__bf16)(acc[mi * 4 + t][v] + bv);
        *(v8bf*)(Vt + (((size_t)(b * 16 + h)) * 64 + d) * 2048 + l0) = r;
      } else {
#pragma unroll
        for (int v = 0; v < 8; v++) {
          int rr = m0 + mi * 16 + v + h8;   // global row in [0,4096)
          int b = rr >> 11, l = rr & 2047;
          float val = acc[mi * 4 + t][v] + bv;
          if (n < 1024) {
            int h = n >> 6, d = n & 63;
            Q[(((size_t)(b * 16 + h)) * 2048 + l) * 64 + d] = (__bf16)val;
          } else {
            int c = n - 1024, h = c >> 6, d = c & 63;
            K[(((size_t)(b * 16 + h)) * 2048 + l) * 64 + d] = (__bf16)val;
          }
        }
      }
    }
  }
}

// Transposed flash attention: compute S^T = K @ Q^T so queries live on lanes and
// keys on VGPR elements. Softmax = in-lane reduction + one shfl_xor(16).
// One wave = 16 query columns, streams 32 keys/iter.
__global__ __launch_bounds__(256) void attn(const __bf16* __restrict__ Q,
                                            const __bf16* __restrict__ K,
                                            const __bf16* __restrict__ Vt,
                                            __bf16* __restrict__ O) {
  __shared__ __bf16 lds[8][16 * 40];   // per-wave P^T staging: [q][k], row stride 40
  const int lane = threadIdx.x & 31;
  const int w    = threadIdx.x >> 5;
  const int qblk = blockIdx.x & 15;
  const int bh   = blockIdx.x >> 4;    // b*16 + h
  const int b = bh >> 4, h = bh & 15;
  const int qbase = qblk * 128 + w * 16;
  const __bf16* Qh = Q + (size_t)bh * 2048 * 64;
  const __bf16* Kh = K + (size_t)bh * 2048 * 64;
  const __bf16* Vh = Vt + (size_t)bh * 64 * 2048;

  const int h8 = (lane >> 4) << 3;
  const int kb = (lane >> 4) << 4;
  const int cl = lane & 15;
  const int q  = qbase + cl;           // this lane's query

  // Q as persistent B fragments (d = 0..31, 32..63)
  v16bf bq0 = load_bt(Qh, 64, qbase, 0, lane);
  v16bf bq1 = load_bt(Qh, 64, qbase, 32, lane);

  float m = -3.0e38f, s = 0.f;
  v8f o[4] = {};                       // o[t]: rows d = t*16 + v + h8, col q

  const int kend = qbase + 16;
  for (int kk = 0; kk < kend; kk += 32) {
    // K rows as A fragments (keys kk..kk+31), V^T rows as A fragments (d rows)
    v16bf ak00 = load_a(Kh, 64, kk, 0, lane);
    v16bf ak01 = load_a(Kh, 64, kk, 32, lane);
    v16bf ak10 = load_a(Kh, 64, kk + 16, 0, lane);
    v16bf ak11 = load_a(Kh, 64, kk + 16, 32, lane);
    v16bf av0 = load_a(Vh, 2048, 0, kk, lane);
    v16bf av1 = load_a(Vh, 2048, 16, kk, lane);
    v16bf av2 = load_a(Vh, 2048, 32, kk, lane);
    v16bf av3 = load_a(Vh, 2048, 48, kk, lane);

    v8f c0 = {}, c1 = {};              // S^T tiles: rows = keys, cols = queries
    c0 = wmma_bf16(ak00, bq0, c0);
    c0 = wmma_bf16(ak01, bq1, c0);
    c1 = wmma_bf16(ak10, bq0, c1);
    c1 = wmma_bf16(ak11, bq1, c1);

    float t0[8], t1[8];
    float mv = -3.0e38f;
#pragma unroll
    for (int v = 0; v < 8; v++) {
      int j0 = kk + v + h8;            // key index of c0[v]
      t0[v] = (j0 <= q)      ? c0[v] * 0.125f : -3.0e38f;
      t1[v] = (j0 + 16 <= q) ? c1[v] * 0.125f : -3.0e38f;
      mv = fmaxf(mv, fmaxf(t0[v], t1[v]));
    }
    mv = fmaxf(mv, __shfl_xor(mv, 16, 32));   // merge key halves
    float mn = fmaxf(m, mv);
    float f = exp2f((m - mn) * L2E);
    m = mn;
    s *= f;
#pragma unroll
    for (int t = 0; t < 4; t++)
#pragma unroll
      for (int v = 0; v < 8; v++) o[t][v] *= f;
    float rs = 0.f;
#pragma unroll
    for (int v = 0; v < 8; v++) {
      t0[v] = exp2f((t0[v] - mn) * L2E);
      t1[v] = exp2f((t1[v] - mn) * L2E);
      rs += t0[v] + t1[v];
    }
    rs += __shfl_xor(rs, 16, 32);
    s += rs;

    // stage P^T into this wave's LDS slice, q-major: lds[q][k_local]
    __bf16* lp = &lds[w][0];
#pragma unroll
    for (int v = 0; v < 8; v++) {
      lp[cl * 40 + (v + h8)]      = (__bf16)t0[v];
      lp[cl * 40 + 16 + (v + h8)] = (__bf16)t1[v];
    }
    // reload as single B fragment: lane col = q, contiguous k = kb..kb+15
    const __bf16* bp = lp + cl * 40 + kb;
    v16bf bP = make16(*(const v8bf*)bp, *(const v8bf*)(bp + 8));

    o[0] = wmma_bf16(av0, bP, o[0]);
    o[1] = wmma_bf16(av1, bP, o[1]);
    o[2] = wmma_bf16(av2, bP, o[2]);
    o[3] = wmma_bf16(av3, bP, o[3]);
  }

  float inv = 1.0f / s;
  // each lane holds 8 consecutive d-values for its query -> 4 coalesced b128 stores
  __bf16* dst = O + ((size_t)(b * 2048 + q)) * 1024 + h * 64 + h8;
#pragma unroll
  for (int t = 0; t < 4; t++) {
    v8bf r;
#pragma unroll
    for (int v = 0; v < 8; v++) r[v] = (__bf16)(o[t][v] * inv);
    *(v8bf*)(dst + t * 16) = r;
  }
}

// out = ab @ wb^T + bias (fp32 result)
__global__ __launch_bounds__(256) void out_gemm(const __bf16* __restrict__ ab,
                                                const __bf16* __restrict__ wb,
                                                const float* __restrict__ bias,
                                                float* __restrict__ out) {
  const int lane = threadIdx.x & 31;
  const int wid  = (blockIdx.x * blockDim.x + threadIdx.x) >> 5;
  const int NG = 1024 / 64;
  const int m0 = (wid / NG) * 32;
  const int n0 = (wid % NG) * 64;
  v8f acc[8] = {};
  gemm_core(ab, wb, m0, n0, lane, acc);

  const int h8 = (lane >> 4) << 3;
#pragma unroll
  for (int mi = 0; mi < 2; mi++) {
#pragma unroll
    for (int t = 0; t < 4; t++) {
      int n = n0 + t * 16 + (lane & 15);
      float bv = bias[n];
#pragma unroll
      for (int v = 0; v < 8; v++) {
        int r = m0 + mi * 16 + v + h8;
        out[(size_t)r * 1024 + n] = acc[mi * 4 + t][v] + bv;
      }
    }
  }
}

extern "C" void kernel_launch(void* const* d_in, const int* in_sizes, int n_in,
                              void* d_out, int out_size, void* d_ws, size_t ws_size,
                              hipStream_t stream) {
  (void)in_sizes; (void)n_in; (void)out_size; (void)ws_size;
  const float* x     = (const float*)d_in[0];
  const float* qkv_w = (const float*)d_in[1];
  const float* qkv_b = (const float*)d_in[2];
  const float* out_w = (const float*)d_in[3];
  const float* out_b = (const float*)d_in[4];
  float* out = (float*)d_out;

  char* ws = (char*)d_ws;
  __bf16* xb   = (__bf16*)ws; ws += (size_t)4096 * 1024 * 2;  // x bf16
  __bf16* wqkv = (__bf16*)ws; ws += (size_t)3072 * 1024 * 2;  // qkv_w bf16
  __bf16* wout = (__bf16*)ws; ws += (size_t)1024 * 1024 * 2;  // out_w bf16
  __bf16* Q    = (__bf16*)ws; ws += (size_t)2 * 16 * 2048 * 64 * 2;
  __bf16* K    = (__bf16*)ws; ws += (size_t)2 * 16 * 2048 * 64 * 2;
  __bf16* Vt   = (__bf16*)ws; ws += (size_t)2 * 16 * 64 * 2048 * 2;
  __bf16* ao   = (__bf16*)ws; ws += (size_t)4096 * 1024 * 2;

  cvt_f32_bf16_v8<<<512, 256, 0, stream>>>(x, xb, 4096 * 1024 / 8);
  cvt_f32_bf16_v8<<<512, 256, 0, stream>>>(qkv_w, wqkv, 3072 * 1024 / 8);
  cvt_f32_bf16_v8<<<512, 256, 0, stream>>>(out_w, wout, 1024 * 1024 / 8);

  // 128 m-tiles(32 rows) * 48 n-groups = 6144 waves -> 768 blocks of 8 waves
  qkv_gemm<<<768, 256, 0, stream>>>(xb, wqkv, qkv_b, Q, K, Vt);

  // (B*H=32) * (L/128=16) = 512 blocks, 8 waves each (16 queries per wave)
  attn<<<512, 256, 0, stream>>>(Q, K, Vt, ao);

  // 128 m-tiles(32 rows) * 16 n-groups = 2048 waves -> 256 blocks
  out_gemm<<<256, 256, 0, stream>>>(ao, wout, out_b, out);
}
